// DecoupleGCN_86844238725530
// MI455X (gfx1250) — matrix operand order repaired and checked by reading, not compiled, not run on hardware
//
#include <hip/hip_runtime.h>
#include <hip/hip_bf16.h>

// ---------------------------------------------------------------------------
// 2-layer GCN: out = A @ (relu(A @ (X @ W1)) @ W2), A as COO (src->dst, vals)
// GEMMs via v_wmma_f32_16x16x32_bf16 (bf16 in, fp32 accum);
// SpMM via float4 gather + hardware global_atomic_add_f32.
// ---------------------------------------------------------------------------

typedef __attribute__((ext_vector_type(16))) __bf16 v16bf;
typedef __attribute__((ext_vector_type(8)))  float  v8f;

#define IN_DIM 128
#define HIDDEN 48
#define NCLS   40

// --------------------------- zero init -------------------------------------
__global__ void zero_f32(float* __restrict__ p, int n) {
  int i = blockIdx.x * blockDim.x + threadIdx.x;
  if (i < n) p[i] = 0.0f;
}

// --------------------- pack W[K,Ncols] into B fragments ---------------------
// One wave (32 threads) per fragment; blockIdx.x = kstep*ntiles + tile.
// B-frag layout (32x16 bf16, wave32): lane L holds col = tile*16 + (L&15),
// khalf = L>>4; frag element i (i = 2v+p, VGPR v bits p) = B[kstep*32 + 8*khalf + i + (i>=8 ? 8 : 0), col].
// Out-of-range (K or Ncols) entries are zero-padded.
__global__ void pack_b_frags(const float* __restrict__ W, unsigned short* __restrict__ pack,
                             int K, int Ncols, int ntiles) {
  int lane  = threadIdx.x;
  int tile  = blockIdx.x % ntiles;
  int kstep = blockIdx.x / ntiles;
  int c = lane & 15, h = lane >> 4;
  int col = tile * 16 + c;
  __bf16* o = (__bf16*)(pack + ((size_t)blockIdx.x * 32 + lane) * 16);
#pragma unroll
  for (int i = 0; i < 16; ++i) {
    int k = kstep * 32 + 8 * h + i + (i >= 8 ? 8 : 0);
    float v = (k < K && col < Ncols) ? W[(size_t)k * Ncols + col] : 0.0f;
    o[i] = (__bf16)v;
  }
}

// --------------------- A-fragment loader (16x32 bf16) ------------------------
// A-frag per ISA: lane row r = L&15, half h = L>>4.
// elements 0..7  = A[r, k0 + 8h + 0..7]      (VGPR0-3)
// elements 8..15 = A[r, k0 + 16 + 8h + 0..7] (VGPR4-7)
// KDIM compile-time so OOB guards fold to zeros (K=48 padding) or vanish (K=128).
template<int KDIM, bool RELU>
__device__ inline v16bf load_a_frag(const float* __restrict__ Ar, int k0, int h) {
  int b1 = k0 + 8 * h;
  int b2 = k0 + 16 + 8 * h;
  float f[16];
#pragma unroll
  for (int i = 0; i < 8; ++i) {
    float x = (b1 + i < KDIM) ? Ar[b1 + i] : 0.0f;
    float y = (b2 + i < KDIM) ? Ar[b2 + i] : 0.0f;
    if (RELU) { x = fmaxf(x, 0.0f); y = fmaxf(y, 0.0f); }
    f[i] = x; f[8 + i] = y;
  }
  v16bf a;
#pragma unroll
  for (int i = 0; i < 16; ++i) a[i] = (__bf16)f[i];
  return a;
}

// ------------------------------ WMMA GEMM ----------------------------------
// C[nRows, ldc] = A[nRows, KDIM] @ Bpack (already in fragment layout).
// One wave per 16-row tile; NTILES 16-col tiles; K fully unrolled.
template<int NTILES, int KDIM, bool RELU>
__global__ void gemm_rowtile_wmma(const float* __restrict__ A,
                                  const unsigned short* __restrict__ Bpack,
                                  float* __restrict__ C,
                                  int nRows, int ldc, int ncols) {
  constexpr int KSTEPS = (KDIM + 31) / 32;
  int wave = threadIdx.x >> 5;
  int lane = threadIdx.x & 31;
  int rowTile = blockIdx.x * (blockDim.x >> 5) + wave;
  int rowBase = rowTile * 16;
  if (rowBase >= nRows) return;

  int r = lane & 15, h = lane >> 4;
  int row = rowBase + r;
  if (row >= nRows) row = nRows - 1;          // clamp (N is a multiple of 16 anyway)
  const float* Arow = A + (size_t)row * KDIM;

  v8f acc[NTILES] = {};
#pragma unroll
  for (int ks = 0; ks < KSTEPS; ++ks) {
    v16bf a = load_a_frag<KDIM, RELU>(Arow, ks * 32, h);
    const unsigned short* bp = Bpack + ((size_t)ks * NTILES * 32 + lane) * 16;
#pragma unroll
    for (int t = 0; t < NTILES; ++t) {
      v16bf b = *(const v16bf*)(bp + (size_t)t * 32 * 16);
      acc[t] = __builtin_amdgcn_wmma_f32_16x16x32_bf16(
          /*neg_a=*/false, a, /*neg_b=*/false, b,
          /*c_mod=*/(short)0, acc[t], /*reuse_a=*/false, /*reuse_b=*/false);
    }
  }

  // C/D layout: lane col n = L&15; VGPR j -> row m = j + 8*(L>>4)
#pragma unroll
  for (int t = 0; t < NTILES; ++t) {
    int col = t * 16 + r;
    if (col < ncols) {
#pragma unroll
      for (int j = 0; j < 8; ++j) {
        int orow = rowBase + 8 * h + j;
        if (orow < nRows) C[(size_t)orow * ldc + col] = acc[t][j];
      }
    }
  }
}

// ------------------------------ COO SpMM -----------------------------------
// out[dst[e], :] += vals[e] * X[src[e], :]; one thread per (edge, 4-dim chunk).
// dst is sorted, so consecutive edges hammer the same L2 lines (cheap atomics).
__global__ void spmm_coo_scatter(const float* __restrict__ X,
                                 const int* __restrict__ src,
                                 const int* __restrict__ dst,
                                 const float* __restrict__ vals,
                                 float* __restrict__ out,
                                 int E, int D) {
  int D4 = D >> 2;
  int idx = blockIdx.x * blockDim.x + threadIdx.x;
  if (idx >= E * D4) return;
  int e = idx / D4;
  int c = (idx - e * D4) << 2;
  float v = vals[e];
  const float4 m = *(const float4*)(X + (size_t)src[e] * D + c);
  float* o = out + (size_t)dst[e] * D + c;
  unsafeAtomicAdd(o + 0, v * m.x);   // global_atomic_add_f32, no return
  unsafeAtomicAdd(o + 1, v * m.y);
  unsafeAtomicAdd(o + 2, v * m.z);
  unsafeAtomicAdd(o + 3, v * m.w);
}

// ------------------------------ launch -------------------------------------
extern "C" void kernel_launch(void* const* d_in, const int* in_sizes, int n_in,
                              void* d_out, int out_size, void* d_ws, size_t ws_size,
                              hipStream_t stream) {
  const float* features = (const float*)d_in[0];   // [N,128]
  const int*   esrc     = (const int*)d_in[1];     // [E]
  const int*   edst     = (const int*)d_in[2];     // [E] (sorted)
  const float* evals    = (const float*)d_in[3];   // [E]
  const float* w1       = (const float*)d_in[4];   // [128,48]
  const float* w2       = (const float*)d_in[5];   // [48,40]
  const int N = in_sizes[0] / IN_DIM;
  const int E = in_sizes[1];
  float* out = (float*)d_out;                      // [N,40]

  // workspace: xw1 [N,48] (later reused as h2 [N,40]) | h [N,48] | packed W1,W2
  size_t seg = ((size_t)N * HIDDEN * sizeof(float) + 255) & ~(size_t)255;
  char* ws = (char*)d_ws;
  float* xw1 = (float*)ws;
  float* h   = (float*)(ws + seg);
  unsigned short* w1p = (unsigned short*)(ws + 2 * seg);            // 4*3 frags * 1KB
  unsigned short* w2p = (unsigned short*)(ws + 2 * seg + 16384);    // 2*3 frags

  const int rowTiles   = (N + 15) / 16;
  const int gemmBlocks = (rowTiles + 3) / 4;       // 4 waves (row tiles) per block

  // pack weights into WMMA B-fragment layout
  pack_b_frags<<<4 * 3, 32, 0, stream>>>(w1, w1p, IN_DIM, HIDDEN, 3);
  pack_b_frags<<<2 * 3, 32, 0, stream>>>(w2, w2p, HIDDEN, NCLS, 3);

  // layer 1: xw1 = X @ W1  (bf16 WMMA, fp32 accum)
  gemm_rowtile_wmma<3, IN_DIM, false><<<gemmBlocks, 128, 0, stream>>>(
      features, w1p, xw1, N, HIDDEN, HIDDEN);

  // h = A @ xw1 (scatter-add)
  int hElems = N * HIDDEN;
  zero_f32<<<(hElems + 255) / 256, 256, 0, stream>>>(h, hElems);
  int total1 = E * (HIDDEN / 4);
  spmm_coo_scatter<<<(total1 + 255) / 256, 256, 0, stream>>>(
      xw1, esrc, edst, evals, h, E, HIDDEN);

  // layer 2: h2 = relu(h) @ W2  (relu fused into A-fragment load); reuse xw1 buf
  gemm_rowtile_wmma<3, HIDDEN, true><<<gemmBlocks, 128, 0, stream>>>(
      h, w2p, xw1, N, NCLS, NCLS);

  // out = A @ h2 (scatter-add)
  zero_f32<<<(N * NCLS + 255) / 256, 256, 0, stream>>>(out, N * NCLS);
  int total2 = E * (NCLS / 4);
  spmm_coo_scatter<<<(total2 + 255) / 256, 256, 0, stream>>>(
      xw1, esrc, edst, evals, out, E, NCLS);
}
